// CustomAttention_7086696039033
// MI455X (gfx1250) — matrix operand order repaired and checked by reading, not compiled, not run on hardware
//
#include <hip/hip_runtime.h>

// ---------------- CDNA5 WMMA types/helpers (wave32) ----------------
typedef __attribute__((ext_vector_type(16))) __bf16 v16bf;
typedef __attribute__((ext_vector_type(8)))  float  v8f;
typedef __attribute__((ext_vector_type(4)))  unsigned int v4u;
typedef __attribute__((ext_vector_type(8)))  int v8i;
typedef __attribute__((ext_vector_type(4)))  int v4i;

union FragBF {
    v16bf v;
    uint4 q[2];
    unsigned int u[8];
};

__device__ __forceinline__ unsigned short f2bf(float f) {
    union { float f; unsigned int u; } x; x.f = f;
    unsigned int r = x.u + 0x7FFFu + ((x.u >> 16) & 1u);   // round-nearest-even
    return (unsigned short)(r >> 16);
}
__device__ __forceinline__ unsigned int pk2(float lo, float hi) {
    return (unsigned int)f2bf(lo) | ((unsigned int)f2bf(hi) << 16);
}
__device__ __forceinline__ v8f wmma_bf16(v16bf a, v16bf b, v8f c) {
    // D = A(16x32 bf16) * B(32x16 bf16) + C(16x16 f32)
    return __builtin_amdgcn_wmma_f32_16x16x32_bf16(false, a, false, b,
                                                   (short)0, c, false, false);
}

// Issue one TDM 2-D tile load (Tensor Data Mover, async, tracked by TENSORcnt).
// D# per CDNA5 ISA ch.8: group0 {count,lds_addr,global_addr,type=2},
// group1 {data_size=2B, tensor dims/strides, tile dims}. Groups 2/3 unused (2-D).
__device__ __forceinline__ void tdm_load_2d(unsigned lds_off, unsigned long long gaddr,
                                            unsigned dim0, unsigned dim1,
                                            unsigned tile0, unsigned tile1,
                                            unsigned long long stride0) {
    v4u g0;
    g0.x = 1u;                                         // count=1 (valid user D#)
    g0.y = lds_off;                                    // lds_addr [63:32]
    g0.z = (unsigned)(gaddr & 0xFFFFFFFFu);            // global_addr lo
    g0.w = (unsigned)((gaddr >> 32) & 0x01FFFFFFu)     // global_addr hi (bits 56:32)
         | (2u << 30);                                 // type=2 ("image")
    v8i g1;
    g1[0] = (int)(1u << 16);                           // workgroup_mask=0, data_size=1 (2B)
    g1[1] = (int)((dim0 & 0xFFFFu) << 16);             // tensor_dim0 lo16 @ [31:16]
    g1[2] = (int)((dim0 >> 16) | ((dim1 & 0xFFFFu) << 16)); // dim0 hi16 | dim1 lo16
    g1[3] = (int)((dim1 >> 16) | (tile0 << 16));       // dim1 hi16 | tile_dim0
    g1[4] = (int)(tile1 & 0xFFFFu);                    // tile_dim1 | tile_dim2=0
    g1[5] = (int)(unsigned)(stride0 & 0xFFFFFFFFu);    // tensor_dim0_stride lo32
    g1[6] = (int)(unsigned)((stride0 >> 32) & 0xFFFFu);// stride0 hi16 | stride1 lo16 = 0
    g1[7] = 0;                                         // stride1 hi = 0
    v4i z = {0, 0, 0, 0};
#if __clang_major__ >= 23
    v8i z8 = {};
    __builtin_amdgcn_tensor_load_to_lds(g0, g1, z, z, z8, 0);
#else
    __builtin_amdgcn_tensor_load_to_lds(g0, g1, z, z, 0);
#endif
}

// ---------------- constants ----------------
#define BATCH 2
#define NSEQ  2048
#define CDIM  1024
#define HEADS 16
#define HD    64
#define C3    3072
#define NTOK  (BATCH * NSEQ)   // 4096

// ---------------- fp32 -> bf16 conversion kernels ----------------
__global__ void conv_pack(const float* __restrict__ src,
                          unsigned int* __restrict__ dst, int n2) {
    int i = blockIdx.x * blockDim.x + threadIdx.x;
    if (i < n2) dst[i] = pk2(src[2 * i], src[2 * i + 1]);
}

// X3[t][c3] = concat(q,k,v)[t][c3] as bf16, token-major [4096][3072]
__global__ void conv_x3(const float* __restrict__ q, const float* __restrict__ k,
                        const float* __restrict__ v, unsigned int* __restrict__ dst) {
    int i = blockIdx.x * blockDim.x + threadIdx.x;
    if (i >= NTOK * C3 / 2) return;
    int e = 2 * i;
    int t = e / C3, c3 = e % C3;
    int which = c3 >> 10, c = c3 & 1023;           // pair never straddles a 1024 boundary
    const float* s = (which == 0) ? q : (which == 1) ? k : v;
    const float* p = s + (size_t)t * CDIM + c;
    dst[i] = pk2(p[0], p[1]);
}

// ---------------- fused QKV projection GEMM (computed transposed) ----------------
// outT = Wqkv(3072x3072 bf16) * X3T ; scatter into Qh/Kh [B,H,N,64] and Vt [B,H,64,N]
__global__ __launch_bounds__(256) void qkv_gemm(
    const unsigned short* __restrict__ X3,   // [4096][3072] bf16
    const unsigned short* __restrict__ Wq,   // [3072][3072] bf16 (row = out channel)
    unsigned short* __restrict__ Qh,
    unsigned short* __restrict__ Kh,
    unsigned short* __restrict__ Vt) {
    const int tid = threadIdx.x, wave = tid >> 5, lane = tid & 31;
    const int ln = lane & 15, kh = lane >> 4;
    const int t0 = blockIdx.x * 64;                       // 4 token subtiles of 16
    const int c3base = (blockIdx.y * 8 + wave) * 16;      // 192 c3-tiles total

    const unsigned short* arow = Wq + (size_t)(c3base + ln) * C3;
    const unsigned short* brow[4];
#pragma unroll
    for (int s = 0; s < 4; ++s) brow[s] = X3 + (size_t)(t0 + s * 16 + ln) * C3;

    v8f acc[4] = {};
    for (int k0 = 0; k0 < C3; k0 += 32) {
        FragBF a;
        a.q[0] = *(const uint4*)(arow + k0 + kh * 16);
        a.q[1] = *(const uint4*)(arow + k0 + kh * 16 + 8);
#pragma unroll
        for (int s = 0; s < 4; ++s) {
            FragBF bfr;
            bfr.q[0] = *(const uint4*)(brow[s] + k0 + kh * 16);
            bfr.q[1] = *(const uint4*)(brow[s] + k0 + kh * 16 + 8);
            acc[s] = wmma_bf16(a.v, bfr.v, acc[s]);
        }
    }

    const int h = (c3base & 1023) >> 6;      // head (uniform per wave)
    const int which = c3base >> 10;          // 0=q 1=k 2=v (uniform per wave)
    const int dstart = (c3base & 63) + kh * 8;
#pragma unroll
    for (int s = 0; s < 4; ++s) {
        const int t = t0 + s * 16 + ln;
        const int b = t >> 11, n = t & 2047;
        if (which < 2) {
            unsigned short* dst = (which == 0 ? Qh : Kh) +
                ((size_t)((b * HEADS + h) * NSEQ + n)) * HD + dstart;
            uint4 pk;
            pk.x = pk2(acc[s][0], acc[s][1]); pk.y = pk2(acc[s][2], acc[s][3]);
            pk.z = pk2(acc[s][4], acc[s][5]); pk.w = pk2(acc[s][6], acc[s][7]);
            *(uint4*)dst = pk;
        } else {
#pragma unroll
            for (int r = 0; r < 8; ++r) {
                const int d = dstart + r;
                Vt[((size_t)((b * HEADS + h) * HD + d)) * NSEQ + n] = f2bf(acc[s][r]);
            }
        }
    }
}

// ---------------- flash attention: TDM-staged K/V in LDS + transposed-S trick ----
__global__ __launch_bounds__(256) void attn_kernel(
    const unsigned short* __restrict__ Qh,   // [B,H,N,64] bf16
    const unsigned short* __restrict__ Kh,   // [B,H,N,64] bf16
    const unsigned short* __restrict__ Vt,   // [B,H,64,N] bf16
    unsigned short* __restrict__ Xa) {       // [4096][1024] bf16
    __shared__ unsigned short bufK[2][32][HD];   // [buf][key][d]   4KB x2
    __shared__ unsigned short bufV[2][HD][32];   // [buf][d][j]     4KB x2

    const int tid = threadIdx.x, wave = tid >> 5, lane = tid & 31;
    const int ln = lane & 15, kh = lane >> 4;
    const int bh = blockIdx.x >> 4;          // 0..31
    const int qb = blockIdx.x & 15;
    const int q0 = qb * 128 + wave * 16;     // 16 query rows per wave
    const int b = bh >> 4, h = bh & 15;

    const unsigned short* Qbase = Qh + (size_t)bh * NSEQ * HD;
    const unsigned long long Kglob = (unsigned long long)(uintptr_t)(Kh) +
                                     (unsigned long long)bh * NSEQ * HD * 2ull;
    const unsigned long long Vglob = (unsigned long long)(uintptr_t)(Vt) +
                                     (unsigned long long)bh * HD * NSEQ * 2ull;

    // Q^T B-fragments: col q = ln, K elems (d) = dh*32 + kh*16 + e (contiguous 32B)
    FragBF bq0, bq1;
    {
        const unsigned short* qrow = Qbase + (size_t)(q0 + ln) * HD;
        bq0.q[0] = *(const uint4*)(qrow + kh * 16);
        bq0.q[1] = *(const uint4*)(qrow + kh * 16 + 8);
        bq1.q[0] = *(const uint4*)(qrow + 32 + kh * 16);
        bq1.q[1] = *(const uint4*)(qrow + 32 + kh * 16 + 8);
    }

    // Prologue: stage key-block 0 (wave 0 drives the Tensor Data Mover)
    if (wave == 0) {
        tdm_load_2d((unsigned)(uintptr_t)&bufK[0][0][0], Kglob,
                    HD, 32, HD, 32, HD);                 // K tile: 32 keys x 64 d
        tdm_load_2d((unsigned)(uintptr_t)&bufV[0][0][0], Vglob,
                    32, HD, 32, HD, NSEQ);               // V tile: 64 d x 32 j
    }

    v8f O[4] = {};
    float m_run = -1e30f, l_run = 0.0f;

    for (int j0 = 0; j0 < NSEQ; j0 += 32) {
        const int cur = (j0 >> 5) & 1;
        if (wave == 0) {
            if (j0 + 32 < NSEQ) {   // prefetch next key block into the other buffer
                const int nxt = cur ^ 1;
                tdm_load_2d((unsigned)(uintptr_t)&bufK[nxt][0][0],
                            Kglob + (unsigned long long)(j0 + 32) * HD * 2ull,
                            HD, 32, HD, 32, HD);
                tdm_load_2d((unsigned)(uintptr_t)&bufV[nxt][0][0],
                            Vglob + (unsigned long long)(j0 + 32) * 2ull,
                            32, HD, 32, HD, NSEQ);
                __builtin_amdgcn_s_wait_tensorcnt(2);    // current block's pair landed
            } else {
                __builtin_amdgcn_s_wait_tensorcnt(0);
            }
        }
        __syncthreads();   // publish bufK/bufV[cur] to all 8 waves

        const unsigned short* bK = &bufK[cur][0][0];
        const unsigned short* bV = &bufV[cur][0][0];

        // --- S^T = K_block * Q^T : two 16-key subtiles, accumulate over d=64 ---
        v8f s0 = {}, s1 = {};
        FragBF a;
        a.q[0] = *(const uint4*)(bK + (ln)*HD + kh * 16);
        a.q[1] = *(const uint4*)(bK + (ln)*HD + kh * 16 + 8);
        s0 = wmma_bf16(a.v, bq0.v, s0);
        a.q[0] = *(const uint4*)(bK + (16 + ln) * HD + kh * 16);
        a.q[1] = *(const uint4*)(bK + (16 + ln) * HD + kh * 16 + 8);
        s1 = wmma_bf16(a.v, bq0.v, s1);
        a.q[0] = *(const uint4*)(bK + (ln)*HD + 32 + kh * 16);
        a.q[1] = *(const uint4*)(bK + (ln)*HD + 32 + kh * 16 + 8);
        s0 = wmma_bf16(a.v, bq1.v, s0);
        a.q[0] = *(const uint4*)(bK + (16 + ln) * HD + 32 + kh * 16);
        a.q[1] = *(const uint4*)(bK + (16 + ln) * HD + 32 + kh * 16 + 8);
        s1 = wmma_bf16(a.v, bq1.v, s1);

        // --- online softmax; lane owns query q = ln (replicated across halves) ---
        float sv[16];
#pragma unroll
        for (int r = 0; r < 8; ++r) { sv[r] = 0.125f * s0[r]; sv[8 + r] = 0.125f * s1[r]; }
        float bm = sv[0];
#pragma unroll
        for (int i = 1; i < 16; ++i) bm = fmaxf(bm, sv[i]);
        bm = fmaxf(bm, __shfl_xor(bm, 16, 32));
        const float mnew  = fmaxf(m_run, bm);
        const float alpha = __expf(m_run - mnew);
        float bsum = 0.0f;
#pragma unroll
        for (int i = 0; i < 16; ++i) { sv[i] = __expf(sv[i] - mnew); bsum += sv[i]; }
        bsum += __shfl_xor(bsum, 16, 32);
        l_run = l_run * alpha + bsum;
        m_run = mnew;
#pragma unroll
        for (int t = 0; t < 4; ++t)
#pragma unroll
            for (int r = 0; r < 8; ++r) O[t][r] *= alpha;

        // --- build P^T B-fragment: cross-half exchange of packed bf16 pairs ---
        unsigned int a0[4], a1[4];
#pragma unroll
        for (int i = 0; i < 4; ++i) {
            a0[i] = pk2(sv[2 * i], sv[2 * i + 1]);
            a1[i] = pk2(sv[8 + 2 * i], sv[9 + 2 * i]);
        }
        FragBF bp;
#pragma unroll
        for (int i = 0; i < 4; ++i) {
            const unsigned int sh0 = __shfl_xor(a0[i], 16, 32);
            const unsigned int sh1 = __shfl_xor(a1[i], 16, 32);
            bp.u[i]     = kh ? sh1   : a0[i];   // K = kh*16 + {0..7}
            bp.u[4 + i] = kh ? a1[i] : sh0;     // K = kh*16 + {8..15}
        }

        // --- O^T += V^T_block * P^T (V tile from LDS) ---
#pragma unroll
        for (int t = 0; t < 4; ++t) {
            FragBF av;
            av.q[0] = *(const uint4*)(bV + (16 * t + ln) * 32 + kh * 16);
            av.q[1] = *(const uint4*)(bV + (16 * t + ln) * 32 + kh * 16 + 8);
            O[t] = wmma_bf16(av.v, bp.v, O[t]);
        }
        __syncthreads();   // all waves done with buf[cur] before TDM overwrites it
    }

    // --- epilogue: normalize, pack to bf16, contiguous 16B stores ---
    const float invl = 1.0f / l_run;
    unsigned short* xrow = Xa + (size_t)(b * NSEQ + q0 + ln) * CDIM + h * HD + kh * 8;
#pragma unroll
    for (int t = 0; t < 4; ++t) {
        uint4 pk;
        pk.x = pk2(O[t][0] * invl, O[t][1] * invl);
        pk.y = pk2(O[t][2] * invl, O[t][3] * invl);
        pk.z = pk2(O[t][4] * invl, O[t][5] * invl);
        pk.w = pk2(O[t][6] * invl, O[t][7] * invl);
        *(uint4*)(xrow + 16 * t) = pk;
    }
}

// ---------------- output projection GEMM (+bias), fp32 out ----------------
__global__ __launch_bounds__(256) void proj_gemm(
    const unsigned short* __restrict__ Xa,   // [4096][1024] bf16
    const unsigned short* __restrict__ Wp,   // [1024][1024] bf16
    const float* __restrict__ bias,          // [1024]
    float* __restrict__ out) {               // [4096][1024] f32
    const int K = CDIM;
    const int tid = threadIdx.x, wave = tid >> 5, lane = tid & 31;
    const int ln = lane & 15, kh = lane >> 4;
    const int t0 = blockIdx.x * 64;
    const int cbase = (blockIdx.y * 8 + wave) * 16;

    const unsigned short* arow = Wp + (size_t)(cbase + ln) * K;
    const unsigned short* brow[4];
#pragma unroll
    for (int s = 0; s < 4; ++s) brow[s] = Xa + (size_t)(t0 + s * 16 + ln) * K;

    v8f acc[4] = {};
    for (int k0 = 0; k0 < K; k0 += 32) {
        FragBF a;
        a.q[0] = *(const uint4*)(arow + k0 + kh * 16);
        a.q[1] = *(const uint4*)(arow + k0 + kh * 16 + 8);
#pragma unroll
        for (int s = 0; s < 4; ++s) {
            FragBF bfr;
            bfr.q[0] = *(const uint4*)(brow[s] + k0 + kh * 16);
            bfr.q[1] = *(const uint4*)(brow[s] + k0 + kh * 16 + 8);
            acc[s] = wmma_bf16(a.v, bfr.v, acc[s]);
        }
    }

    float bv[8];
#pragma unroll
    for (int r = 0; r < 8; ++r) bv[r] = bias[cbase + kh * 8 + r];
#pragma unroll
    for (int s = 0; s < 4; ++s) {
        const int t = t0 + s * 16 + ln;
        float* dst = out + (size_t)t * CDIM + cbase + kh * 8;
        float4 lo = { acc[s][0] + bv[0], acc[s][1] + bv[1], acc[s][2] + bv[2], acc[s][3] + bv[3] };
        float4 hi = { acc[s][4] + bv[4], acc[s][5] + bv[5], acc[s][6] + bv[6], acc[s][7] + bv[7] };
        *(float4*)dst = lo;
        *(float4*)(dst + 4) = hi;
    }
}

// ---------------- launch ----------------
extern "C" void kernel_launch(void* const* d_in, const int* in_sizes, int n_in,
                              void* d_out, int out_size, void* d_ws, size_t ws_size,
                              hipStream_t stream) {
    (void)in_sizes; (void)n_in; (void)out_size; (void)ws_size;
    const float* q     = (const float*)d_in[0];
    const float* k     = (const float*)d_in[1];
    const float* v     = (const float*)d_in[2];
    const float* Wqkv  = (const float*)d_in[3];
    const float* Wproj = (const float*)d_in[4];
    const float* bproj = (const float*)d_in[5];
    float* out = (float*)d_out;

    char* ws = (char*)d_ws;
    size_t off = 0;
    auto alloc = [&](size_t bytes) -> void* {
        void* p = ws + off;
        off += (bytes + 255) & ~(size_t)255;
        return p;
    };
    unsigned short* X3 = (unsigned short*)alloc((size_t)NTOK * C3 * 2);      // 25.2 MB
    unsigned short* Wq = (unsigned short*)alloc((size_t)C3 * C3 * 2);        // 18.9 MB
    unsigned short* Wp = (unsigned short*)alloc((size_t)CDIM * CDIM * 2);    //  2.1 MB
    unsigned short* Qh = (unsigned short*)alloc((size_t)NTOK * CDIM * 2);    //  8.4 MB
    unsigned short* Kh = (unsigned short*)alloc((size_t)NTOK * CDIM * 2);    //  8.4 MB
    unsigned short* Vt = (unsigned short*)alloc((size_t)NTOK * CDIM * 2);    //  8.4 MB
    unsigned short* Xa = (unsigned short*)alloc((size_t)NTOK * CDIM * 2);    //  8.4 MB

    {   // conversions
        const int n2x = NTOK * C3 / 2;
        conv_x3<<<(n2x + 255) / 256, 256, 0, stream>>>(q, k, v, (unsigned int*)X3);
        const int n2w = C3 * C3 / 2;
        conv_pack<<<(n2w + 255) / 256, 256, 0, stream>>>(Wqkv, (unsigned int*)Wq, n2w);
        const int n2p = CDIM * CDIM / 2;
        conv_pack<<<(n2p + 255) / 256, 256, 0, stream>>>(Wproj, (unsigned int*)Wp, n2p);
    }
    // QKV projection: 192 c3-tiles x 64 token-tiles, 8 waves/WG
    qkv_gemm<<<dim3(64, 24), 256, 0, stream>>>(X3, Wq, Qh, Kh, Vt);
    // attention: one wave per 16-query tile; 32 (b,h) x 16 q-blocks; TDM-staged K/V
    attn_kernel<<<512, 256, 0, stream>>>(Qh, Kh, Vt, Xa);
    // output projection: 64 c-tiles x 64 token-tiles
    proj_gemm<<<dim3(64, 8), 256, 0, stream>>>(Xa, Wp, bproj, out);
}
